// PointNet_11862699671553
// MI455X (gfx1250) — compile-verified, matrix operand order checked by reading, not run on hardware
//
#include <hip/hip_runtime.h>

typedef _Float16 f16;
typedef __attribute__((ext_vector_type(16))) _Float16 v16h;
typedef __attribute__((ext_vector_type(8)))  float    v8f;

// ---------------------------------------------------------------------------
// CDNA5 v_wmma_f32_16x16x32_f16 register swizzles (ISA 7.12.2, wave32).
// A (16x32): lanes 0-15 row M=lane with K={0..7,16..23}; lanes 16-31 row
// M=lane-16 with K={8..15,24..31}. B (32x16) mirrors with lanes indexing N.
// Inverse map for a K-offset k32 in [0,32):
//   laneHalf = (k32>>3)&1      (0 -> lanes 0-15, 1 -> lanes 16-31)
//   halfIdx  = (k32&7) | ((k32>>1)&8)
// C/D (16x16 f32): VGPR r -> (M=r, N=lane) / (M=r+8, N=lane-16).
// Activations are staged in LDS *in A-layout* ([mt][kt][lane][16]) and weights
// are pre-packed to f16 in B-layout ([kt][nt][lane][16]) so the inner loop is
// pure b128 loads + wmma.
// ---------------------------------------------------------------------------
__device__ __forceinline__ int kmap16(int i, int hi) {
    return hi ? ((i < 8) ? (i + 8) : (i + 16))
              : ((i < 8) ? i       : (i + 8));
}

// Scatter one logical element (row R, col c) into an A-swizzled LDS buffer.
__device__ __forceinline__ void store_a_elem(v16h* Xa, int KT, int R, int c, float v) {
    int mt = R >> 4, r = R & 15, kt = c >> 5, k32 = c & 31;
    int lane = r + (((k32 >> 3) & 1) << 4);
    int i    = (k32 & 7) | ((k32 >> 1) & 8);
    ((f16*)&Xa[(mt * KT + kt) * 32 + lane])[i] = (f16)v;
}

// ---------------------------------------------------------------------------
// Weight pre-pack: W fp32 [CIN][COUT] -> f16 B-swizzled tiles, K zero-padded.
// Wp flat index = ((kt*NT + nt)*32 + lane)*16 + i.
// ---------------------------------------------------------------------------
__global__ void pack_w_kernel(const float* __restrict__ W, f16* __restrict__ Wp,
                              int CIN, int COUT, int total) {
    int t = blockIdx.x * 256 + threadIdx.x;
    if (t >= total) return;
    int i    = t & 15;
    int lane = (t >> 4) & 31;
    int tile = t >> 9;
    int NT = COUT >> 4;
    int kt = tile / NT, nt = tile - kt * NT;
    int k = kt * 32 + kmap16(i, lane >> 4);
    int n = nt * 16 + (lane & 15);
    Wp[t] = (k < CIN) ? (f16)W[k * COUT + n] : (f16)0.f;
}

// ---------------------------------------------------------------------------
// One MLP layer: H = relu((X @ W) * g + b). A from LDS (A-layout, 2x
// ds_load_b128), B from global f16 pack (2x global_load_b128), f32 WMMA
// accumulate, output scattered back to LDS in A-layout for the next layer.
// ---------------------------------------------------------------------------
template<int KTIN, int COUT>
__device__ __forceinline__ void mlp_layer(const v16h* __restrict__ Xa,
                                          const v16h* __restrict__ Wp,
                                          const float* __restrict__ gam,
                                          const float* __restrict__ bet,
                                          v16h* __restrict__ Xout,
                                          int mt, int nt0, int ntStep) {
    constexpr int NT = COUT / 16;
    constexpr int KTOUT = COUT / 32;
    int lane = threadIdx.x & 31;
    int hi   = lane >> 4;
    int nl   = lane & 15;
    for (int nt = nt0; nt < NT; nt += ntStep) {
        v8f acc = {};
#pragma unroll
        for (int kt = 0; kt < KTIN; ++kt) {
            v16h a = Xa[(mt * KTIN + kt) * 32 + lane];
            v16h b = Wp[(kt * NT + nt) * 32 + lane];
            acc = __builtin_amdgcn_wmma_f32_16x16x32_f16(false, a, false, b,
                                                         (short)0, acc, false, false);
        }
        int n = nt * 16 + nl;
        float gg = gam[n], bb = bet[n];
        int k32 = n & 31, ktn = n >> 5;
        int laneHiOff = ((k32 >> 3) & 1) << 4;
        int i2 = (k32 & 7) | ((k32 >> 1) & 8);
        f16* dst = (f16*)&Xout[(mt * KTOUT + ktn) * 32] + i2;
#pragma unroll
        for (int r = 0; r < 8; ++r) {
            float v = acc[r] * gg + bb;
            v = v > 0.f ? v : 0.f;
            dst[(r + hi * 8 + laneHiOff) * 16] = (f16)v;
        }
    }
}

// Final layer fused with max over the wave's 16 rows -> per-wave partials in LDS.
template<int KTIN, int COUT>
__device__ __forceinline__ void mlp_layer_maxred(const v16h* __restrict__ Xa,
                                                 const v16h* __restrict__ Wp,
                                                 const float* __restrict__ gam,
                                                 const float* __restrict__ bet,
                                                 float* __restrict__ red,
                                                 int mt, int wave) {
    constexpr int NT = COUT / 16;
    int lane = threadIdx.x & 31;
    int nl   = lane & 15;
    for (int nt = 0; nt < NT; ++nt) {
        v8f acc = {};
#pragma unroll
        for (int kt = 0; kt < KTIN; ++kt) {
            v16h a = Xa[(mt * KTIN + kt) * 32 + lane];
            v16h b = Wp[(kt * NT + nt) * 32 + lane];
            acc = __builtin_amdgcn_wmma_f32_16x16x32_f16(false, a, false, b,
                                                         (short)0, acc, false, false);
        }
        int n = nt * 16 + nl;
        float gg = gam[n], bb = bet[n];
        float mx = 0.f;                         // post-ReLU values are >= 0
#pragma unroll
        for (int r = 0; r < 8; ++r) {
            float v = acc[r] * gg + bb;
            v = v > 0.f ? v : 0.f;
            mx = fmaxf(mx, v);
        }
        mx = fmaxf(mx, __shfl_xor(mx, 16, 32)); // merge M=r with M=r+8 halves
        if (lane < 16) red[wave * COUT + n] = mx;
    }
}

// Final layer fused with max over 16 rows + bit-pattern atomicMax to global.
template<int KTIN, int COUT>
__device__ __forceinline__ void mlp_layer_maxatomic(const v16h* __restrict__ Xa,
                                                    const v16h* __restrict__ Wp,
                                                    const float* __restrict__ gam,
                                                    const float* __restrict__ bet,
                                                    int* __restrict__ gout,
                                                    int nt0, int ntStep) {
    constexpr int NT = COUT / 16;
    int lane = threadIdx.x & 31;
    int nl   = lane & 15;
    for (int nt = nt0; nt < NT; nt += ntStep) {
        v8f acc = {};
#pragma unroll
        for (int kt = 0; kt < KTIN; ++kt) {
            v16h a = Xa[kt * 32 + lane];
            v16h b = Wp[(kt * NT + nt) * 32 + lane];
            acc = __builtin_amdgcn_wmma_f32_16x16x32_f16(false, a, false, b,
                                                         (short)0, acc, false, false);
        }
        int n = nt * 16 + nl;
        float gg = gam[n], bb = bet[n];
        float mx = 0.f;
#pragma unroll
        for (int r = 0; r < 8; ++r) {
            float v = acc[r] * gg + bb;
            v = v > 0.f ? v : 0.f;
            mx = fmaxf(mx, v);
        }
        mx = fmaxf(mx, __shfl_xor(mx, 16, 32));
        if (lane < 16) atomicMax(&gout[n], __float_as_int(mx)); // exact for v >= 0
    }
}

// ---------------------------------------------------------------------------
// Simple gather: dst[b,m,:] = src[b, idx[b,m], :] (3 floats per point)
// ---------------------------------------------------------------------------
__global__ void gather3_kernel(const float* __restrict__ src, const int* __restrict__ idx,
                               float* __restrict__ dst, int nSrc, int M, int total) {
    int t = blockIdx.x * blockDim.x + threadIdx.x;
    if (t >= total) return;
    int b = t / M;
    int j = idx[t];
    const float* s = src + ((long)b * nSrc + j) * 3;
    float* d = dst + (long)t * 3;
    d[0] = s[0]; d[1] = s[1]; d[2] = s[2];
}

__global__ void zero_kernel(float* __restrict__ p, int n) {
    int t = blockIdx.x * blockDim.x + threadIdx.x;
    if (t < n) p[t] = 0.f;
}

// ---------------------------------------------------------------------------
// KNN: one 128-thread block per query; distances in LDS; 64 argmin passes.
// ---------------------------------------------------------------------------
template<int N>
__global__ __launch_bounds__(128) void knn_kernel(const float* __restrict__ src,
                                                  const float* __restrict__ qry,
                                                  int* __restrict__ nidx, int M) {
    __shared__ float d2[N];
    __shared__ float rv[128];
    __shared__ int   ri[128];
    int b = blockIdx.y, m = blockIdx.x, tid = threadIdx.x;
    const float* sp = src + (long)b * N * 3;
    const float* qp = qry + ((long)b * M + m) * 3;
    float qx = qp[0], qy = qp[1], qz = qp[2];
    for (int j = tid; j < N; j += 128) {
        float dx = sp[3 * j + 0] - qx;
        float dy = sp[3 * j + 1] - qy;
        float dz = sp[3 * j + 2] - qz;
        d2[j] = dx * dx + dy * dy + dz * dz;
    }
    __syncthreads();
    int* out = nidx + ((long)b * M + m) * 64;
    for (int t = 0; t < 64; ++t) {
        float best = 3.4e38f; int bi = 0x7fffffff;
        for (int j = tid; j < N; j += 128) {
            float v = d2[j];
            if (v < best) { best = v; bi = j; }
        }
        rv[tid] = best; ri[tid] = bi;
        __syncthreads();
        for (int s = 64; s > 0; s >>= 1) {
            if (tid < s) {
                float ov = rv[tid + s]; int oi = ri[tid + s];
                if (ov < rv[tid] || (ov == rv[tid] && oi < ri[tid])) {
                    rv[tid] = ov; ri[tid] = oi;
                }
            }
            __syncthreads();
        }
        if (tid == 0) { out[t] = ri[0]; d2[ri[0]] = 3.4e38f; }
        __syncthreads();
    }
}

// ---------------------------------------------------------------------------
// Point conv: one block per (b, centroid). 64 neighbors -> 4 M-tiles, one per
// wave. 3-layer WMMA MLP, activations staged in LDS in A-layout, final layer
// fused with max over the 64 neighbors.
// ---------------------------------------------------------------------------
template<int CFEAT, int CIN, int CINP, int C1, int C2, int C3>
__global__ __launch_bounds__(128) void point_conv_kernel(
    const float* __restrict__ src_pos, const float* __restrict__ feat_in,
    const float* __restrict__ q_pos,  const int* __restrict__ nidx,
    const v16h* __restrict__ Wp1, const float* __restrict__ g1, const float* __restrict__ b1,
    const v16h* __restrict__ Wp2, const float* __restrict__ g2, const float* __restrict__ b2,
    const v16h* __restrict__ Wp3, const float* __restrict__ g3, const float* __restrict__ b3,
    float* __restrict__ feat_out, int nsrc, int M) {
    constexpr int KT0 = CINP / 32, KT1 = C1 / 32, KT2 = C2 / 32;
    __shared__ v16h X0a[4 * KT0 * 32];
    __shared__ v16h X1a[4 * KT1 * 32];
    __shared__ v16h X2a[4 * KT2 * 32];
    __shared__ float red[4 * C3];
    int b = blockIdx.y, m = blockIdx.x;
    int tid = threadIdx.x, wave = tid >> 5;
    long base = (long)b * M + m;
    const int* nn = nidx + base * 64;
    const float* ctr = q_pos + base * 3;
    float cx = ctr[0], cy = ctr[1], cz = ctr[2];
    // Stage [64 neighbors x CINP]: [recentered pos(3) | feat(CFEAT) | zero pad]
    for (int e = tid; e < 64 * CINP; e += 128) {
        int R = e / CINP, c = e - R * CINP;
        int j = nn[R];
        float v = 0.f;
        if (c < 3) {
            float p = src_pos[((long)b * nsrc + j) * 3 + c];
            v = p - (c == 0 ? cx : (c == 1 ? cy : cz));
        } else if (c < CIN) {
            v = feat_in[((long)b * nsrc + j) * CFEAT + (c - 3)];
        }
        store_a_elem(X0a, KT0, R, c, v);
    }
    __syncthreads();
    int mt = wave;
    mlp_layer<KT0, C1>(X0a, Wp1, g1, b1, X1a, mt, 0, 1);
    __syncthreads();
    mlp_layer<KT1, C2>(X1a, Wp2, g2, b2, X2a, mt, 0, 1);
    __syncthreads();
    mlp_layer_maxred<KT2, C3>(X2a, Wp3, g3, b3, red, mt, wave);
    __syncthreads();
    for (int n = tid; n < C3; n += 128) {
        float v = fmaxf(fmaxf(red[n], red[C3 + n]),
                        fmaxf(red[2 * C3 + n], red[3 * C3 + n]));
        feat_out[base * C3 + n] = v;
    }
}

// ---------------------------------------------------------------------------
// Global MLP: one block per (b, 16-point tile); waves split N-tiles; final
// layer max-pools over the 16 rows and atomicMax-merges into gmax[b,1024].
// ---------------------------------------------------------------------------
template<int CF, int CINP, int C1, int C2, int C3>
__global__ __launch_bounds__(128) void glob_kernel(
    const float* __restrict__ feat, const float* __restrict__ pos,
    const v16h* __restrict__ Wp1, const float* __restrict__ g1, const float* __restrict__ b1,
    const v16h* __restrict__ Wp2, const float* __restrict__ g2, const float* __restrict__ b2,
    const v16h* __restrict__ Wp3, const float* __restrict__ g3, const float* __restrict__ b3,
    int* __restrict__ gout, int M) {
    constexpr int KT0 = CINP / 32, KT1 = C1 / 32, KT2 = C2 / 32;
    __shared__ v16h X0a[KT0 * 32];
    __shared__ v16h X1a[KT1 * 32];
    __shared__ v16h X2a[KT2 * 32];
    int b = blockIdx.y, pt = blockIdx.x;
    int tid = threadIdx.x, wave = tid >> 5;
    int p0 = pt * 16;
    for (int e = tid; e < 16 * CINP; e += 128) {
        int R = e / CINP, c = e - R * CINP;
        long p = (long)b * M + p0 + R;
        float v = 0.f;
        if (c < CF)          v = feat[p * CF + c];
        else if (c < CF + 3) v = pos[p * 3 + (c - CF)];
        store_a_elem(X0a, KT0, R, c, v);
    }
    __syncthreads();
    mlp_layer<KT0, C1>(X0a, Wp1, g1, b1, X1a, 0, wave, 4);
    __syncthreads();
    mlp_layer<KT1, C2>(X1a, Wp2, g2, b2, X2a, 0, wave, 4);
    __syncthreads();
    mlp_layer_maxatomic<KT2, C3>(X2a, Wp3, g3, b3, gout + (long)b * C3, wave, 4);
}

// ---------------------------------------------------------------------------
// FC head: tiny (8 rows), plain VALU.
// ---------------------------------------------------------------------------
__global__ __launch_bounds__(256) void fc_kernel(
    const float* __restrict__ x,  // [8,1024]
    const float* __restrict__ W1, const float* __restrict__ g1, const float* __restrict__ b1,
    const float* __restrict__ W2, const float* __restrict__ g2, const float* __restrict__ b2,
    const float* __restrict__ Wo, const float* __restrict__ bo,
    float* __restrict__ out) {
    __shared__ float h1[8 * 512];
    __shared__ float h2[8 * 256];
    int tid = threadIdx.x;
    for (int e = tid; e < 8 * 512; e += 256) {
        int r = e >> 9, n = e & 511;
        float s = 0.f;
        for (int k = 0; k < 1024; ++k) s += x[r * 1024 + k] * W1[k * 512 + n];
        float v = s * g1[n] + b1[n];
        h1[e] = v > 0.f ? v : 0.f;
    }
    __syncthreads();
    for (int e = tid; e < 8 * 256; e += 256) {
        int r = e >> 8, n = e & 255;
        float s = 0.f;
        for (int k = 0; k < 512; ++k) s += h1[r * 512 + k] * W2[k * 256 + n];
        float v = s * g2[n] + b2[n];
        h2[e] = v > 0.f ? v : 0.f;
    }
    __syncthreads();
    for (int e = tid; e < 8 * 40; e += 256) {
        int r = e / 40, n = e - r * 40;
        float s = 0.f;
        for (int k = 0; k < 256; ++k) s += h2[r * 256 + k] * Wo[k * 40 + n];
        out[e] = s + bo[n];
    }
}

// ---------------------------------------------------------------------------
// Host-side pipeline
// ---------------------------------------------------------------------------
extern "C" void kernel_launch(void* const* d_in, const int* in_sizes, int n_in,
                              void* d_out, int out_size, void* d_ws, size_t ws_size,
                              hipStream_t stream) {
    (void)in_sizes; (void)n_in; (void)out_size; (void)ws_size;
    const int B = 8, NALL = 8192, N0 = 4096, M1 = 1024, M2 = 256;

    const float* all_points = (const float*)d_in[0];
    const int*   idx0       = (const int*)d_in[1];
    const int*   idx1       = (const int*)d_in[2];
    const int*   idx2       = (const int*)d_in[3];
    const float* c1[9]; for (int i = 0; i < 9; ++i) c1[i] = (const float*)d_in[4 + i];
    const float* c2[9]; for (int i = 0; i < 9; ++i) c2[i] = (const float*)d_in[13 + i];
    const float* gl[9]; for (int i = 0; i < 9; ++i) gl[i] = (const float*)d_in[22 + i];
    const float* fc[6]; for (int i = 0; i < 6; ++i) fc[i] = (const float*)d_in[31 + i];
    const float* foW = (const float*)d_in[37];
    const float* fob = (const float*)d_in[38];

    char* ws = (char*)d_ws;
    size_t off = 0;
    auto take = [&](size_t bytes) -> char* {
        char* p = ws + off;
        off += (bytes + 255) & ~(size_t)255;
        return p;
    };
    float* pos0  = (float*)take((size_t)B * N0 * 3 * 4);
    float* pos1  = (float*)take((size_t)B * M1 * 3 * 4);
    float* pos2  = (float*)take((size_t)B * M2 * 3 * 4);
    int*   nidx1 = (int*)  take((size_t)B * M1 * 64 * 4);
    int*   nidx2 = (int*)  take((size_t)B * M2 * 64 * 4);
    float* feat1 = (float*)take((size_t)B * M1 * 128 * 4);
    float* feat2 = (float*)take((size_t)B * M2 * 256 * 4);
    float* gmax  = (float*)take((size_t)B * 1024 * 4);

    // Pre-pack all conv/glob weights into f16 B-swizzled tiles (L2-resident).
    auto pack = [&](const float* W, int CIN, int COUT, int KT) -> v16h* {
        int total = KT * (COUT / 16) * 512;
        f16* dst = (f16*)take((size_t)total * sizeof(f16));
        pack_w_kernel<<<(total + 255) / 256, 256, 0, stream>>>(W, dst, CIN, COUT, total);
        return (v16h*)dst;
    };
    v16h* p1a = pack(c1[0],   6,   64, 1);
    v16h* p1b = pack(c1[3],  64,   64, 2);
    v16h* p1c = pack(c1[6],  64,  128, 2);
    v16h* p2a = pack(c2[0], 131,  128, 5);
    v16h* p2b = pack(c2[3], 128,  128, 4);
    v16h* p2c = pack(c2[6], 128,  256, 4);
    v16h* pga = pack(gl[0], 259,  256, 9);
    v16h* pgb = pack(gl[3], 256,  512, 8);
    v16h* pgc = pack(gl[6], 512, 1024, 16);

    // Stage 0: index gathers
    gather3_kernel<<<(B * N0 + 255) / 256, 256, 0, stream>>>(all_points, idx0, pos0, NALL, N0, B * N0);
    gather3_kernel<<<(B * M1 + 255) / 256, 256, 0, stream>>>(pos0, idx1, pos1, N0, M1, B * M1);
    gather3_kernel<<<(B * M2 + 255) / 256, 256, 0, stream>>>(pos1, idx2, pos2, M1, M2, B * M2);

    // Stage 1: KNN + grouped MLP (6 -> 64 -> 64 -> 128), max over 64 neighbors
    knn_kernel<4096><<<dim3(M1, B), 128, 0, stream>>>(pos0, pos1, nidx1, M1);
    point_conv_kernel<3, 6, 32, 64, 64, 128><<<dim3(M1, B), 128, 0, stream>>>(
        pos0, pos0, pos1, nidx1,
        p1a, c1[1], c1[2], p1b, c1[4], c1[5], p1c, c1[7], c1[8],
        feat1, N0, M1);

    // Stage 2: KNN + grouped MLP (131 -> 128 -> 128 -> 256)
    knn_kernel<1024><<<dim3(M2, B), 128, 0, stream>>>(pos1, pos2, nidx2, M2);
    point_conv_kernel<128, 131, 160, 128, 128, 256><<<dim3(M2, B), 128, 0, stream>>>(
        pos1, feat1, pos2, nidx2,
        p2a, c2[1], c2[2], p2b, c2[4], c2[5], p2c, c2[7], c2[8],
        feat2, M1, M2);

    // Stage 3: global MLP (259 -> 256 -> 512 -> 1024) + global max pool
    zero_kernel<<<(B * 1024 + 255) / 256, 256, 0, stream>>>(gmax, B * 1024);
    glob_kernel<256, 288, 256, 512, 1024><<<dim3(M2 / 16, B), 128, 0, stream>>>(
        feat2, pos2,
        pga, gl[1], gl[2], pgb, gl[4], gl[5], pgc, gl[7], gl[8],
        (int*)gmax, M2);

    // Stage 4: FC head (1024 -> 512 -> 256 -> 40)
    fc_kernel<<<1, 256, 0, stream>>>(gmax,
        fc[0], fc[1], fc[2], fc[3], fc[4], fc[5],
        foW, fob, (float*)d_out);
}
